// Decoder_3126736191506
// MI455X (gfx1250) — compile-verified, matrix operand order checked by reading, not run on hardware
//
#include <hip/hip_runtime.h>
#include <hip/hip_bf16.h>

// ---------------------------------------------------------------------------
// CDNA5 (gfx1250) transformer decoder, bf16-WMMA GEMM core, wave32.
// Double-buffered LDS staging: one barrier per K-step, global loads for tile
// k+1 overlap WMMA on tile k; global_prefetch for tile k+2.
// ---------------------------------------------------------------------------

typedef __attribute__((ext_vector_type(16))) __bf16 v16bf;
typedef __attribute__((ext_vector_type(8)))  __bf16 v8bf;
typedef __attribute__((ext_vector_type(8)))  float  v8f;

#define BM 128
#define BN 128
#define BK 32

#define D_EMB 2048
#define D_FF  8192
#define NTOK  1024   // BATCH * SEQ
#define SEQL  128
#define NHEAD 16
#define DK    128
#define VOCAB 30000

// Load 16 consecutive f32 and convert to 2x v8bf.
__device__ __forceinline__ void stage16(const float* __restrict__ src,
                                        v8bf& d0, v8bf& d1)
{
    const float4* s4 = reinterpret_cast<const float4*>(src);
    float f[16];
#pragma unroll
    for (int i = 0; i < 4; i++) {
        float4 t4 = s4[i];
        f[4*i+0] = t4.x; f[4*i+1] = t4.y; f[4*i+2] = t4.z; f[4*i+3] = t4.w;
    }
#pragma unroll
    for (int i = 0; i < 8; i++) { d0[i] = (__bf16)f[i]; d1[i] = (__bf16)f[i+8]; }
}

// ---------------------------------------------------------------------------
// GEMM: C[M,N] = A[M,K] (f32, row-major) * B[N,K]^T (f32, row-major, weights)
//       + optional bias[N], optional ReLU.  M % 128 == 0, K % 32 == 0.
// f32 -> bf16 conversion happens while staging into LDS; WMMA accumulates f32.
// ---------------------------------------------------------------------------
__global__ __launch_bounds__(256) void gemm_wmma_bf16(
    const float* __restrict__ A, const float* __restrict__ B,
    const float* __restrict__ bias, float* __restrict__ C,
    int M, int N, int K, int relu)
{
    // ping-pong buffers
    __shared__ __attribute__((aligned(16))) __bf16 As[2 * BM * BK];
    __shared__ __attribute__((aligned(16))) __bf16 Bs[2 * BM * BK];

    const int tid  = threadIdx.x;
    const int wave = tid >> 5;        // 0..7
    const int lane = tid & 31;
    const int h    = lane >> 4;       // lane half: 0/1
    const int lm   = lane & 15;

    const int wm = wave & 3;          // 4 wave-rows of 32
    const int wn = wave >> 2;         // 2 wave-cols of 64

    const int blockM = blockIdx.y * BM;
    const int blockN = blockIdx.x * BN;

    // Staging map: thread t loads one 16-float run: row = t>>1, col = (t&1)*16
    const int ldRow = tid >> 1;
    const int ldCol = (tid & 1) * 16;
    const int bRowN = blockN + ldRow;                 // B row (guard vs N)
    const float* aSrc = A + (size_t)(blockM + ldRow) * K + ldCol;
    const float* bSrc = B + (size_t)bRowN * K + ldCol;

    v8f acc[2][4];
#pragma unroll
    for (int t = 0; t < 2; t++)
#pragma unroll
        for (int u = 0; u < 4; u++)
#pragma unroll
            for (int r = 0; r < 8; r++) acc[t][u][r] = 0.0f;

    // ---- prologue: stage tile 0 into registers ----
    v8bf a0, a1, b0, b1;
    stage16(aSrc, a0, a1);
    if (bRowN < N) {
        stage16(bSrc, b0, b1);
    } else {
#pragma unroll
        for (int i = 0; i < 8; i++) { b0[i] = (__bf16)0.0f; b1[i] = (__bf16)0.0f; }
    }

    int p = 0;
    for (int k0 = 0; k0 < K; k0 += BK) {
        // ---- commit staged registers to LDS buffer p ----
        __bf16* Asp = As + p * (BM * BK);
        __bf16* Bsp = Bs + p * (BM * BK);
        *reinterpret_cast<v8bf*>(&Asp[ldRow * BK + ldCol + 0]) = a0;
        *reinterpret_cast<v8bf*>(&Asp[ldRow * BK + ldCol + 8]) = a1;
        *reinterpret_cast<v8bf*>(&Bsp[ldRow * BK + ldCol + 0]) = b0;
        *reinterpret_cast<v8bf*>(&Bsp[ldRow * BK + ldCol + 8]) = b1;
        __syncthreads();

        // ---- issue long-latency global loads for tile k+1 (overlaps WMMA) ----
        const int k1 = k0 + BK;
        if (k1 < K) {
            stage16(aSrc + k1, a0, a1);
            if (bRowN < N) {
                stage16(bSrc + k1, b0, b1);
            } else {
#pragma unroll
                for (int i = 0; i < 8; i++) { b0[i] = (__bf16)0.0f; b1[i] = (__bf16)0.0f; }
            }
            // prefetch tile k+2 toward L2/WGP$ (global_prefetch_b8)
            const int k2 = k1 + BK;
            if (k2 < K) {
                __builtin_prefetch(aSrc + k2, 0, 1);
                if (bRowN < N) __builtin_prefetch(bSrc + k2, 0, 1);
            }
        }

        // ---- fragments from LDS buffer p (two ds_load_b128 each) ----
        // A 16-bit layout: lane half h holds K = 8h+0..7 (elems 0..7) and
        // 16+8h+0..7 (elems 8..15); B layout mirrors with M->N.
        v16bf aF[2], bF[4];
#pragma unroll
        for (int t = 0; t < 2; t++) {
            const int row = wm * 32 + t * 16 + lm;
            const v8bf lo = *reinterpret_cast<const v8bf*>(&Asp[row * BK + 8 * h]);
            const v8bf hi = *reinterpret_cast<const v8bf*>(&Asp[row * BK + 16 + 8 * h]);
            aF[t] = __builtin_shufflevector(lo, hi, 0,1,2,3,4,5,6,7,8,9,10,11,12,13,14,15);
        }
#pragma unroll
        for (int u = 0; u < 4; u++) {
            const int col = wn * 64 + u * 16 + lm;
            const v8bf lo = *reinterpret_cast<const v8bf*>(&Bsp[col * BK + 8 * h]);
            const v8bf hi = *reinterpret_cast<const v8bf*>(&Bsp[col * BK + 16 + 8 * h]);
            bF[u] = __builtin_shufflevector(lo, hi, 0,1,2,3,4,5,6,7,8,9,10,11,12,13,14,15);
        }

#pragma unroll
        for (int t = 0; t < 2; t++)
#pragma unroll
            for (int u = 0; u < 4; u++)
                acc[t][u] = __builtin_amdgcn_wmma_f32_16x16x32_bf16(
                    false, aF[t], false, bF[u], (short)0, acc[t][u], false, false);

        p ^= 1;   // buffer p is safely rewritten only after the next barrier
    }

    // ---- write C: VGPR r, lane l -> tile row = r + 8h, tile col = lm ----
#pragma unroll
    for (int t = 0; t < 2; t++) {
        const int rowBase = blockM + wm * 32 + t * 16 + 8 * h;
#pragma unroll
        for (int u = 0; u < 4; u++) {
            const int col = blockN + wn * 64 + u * 16 + lm;
            if (col < N) {
                const float bv = bias ? bias[col] : 0.0f;
#pragma unroll
                for (int r = 0; r < 8; r++) {
                    float v = acc[t][u][r] + bv;
                    if (relu) v = v > 0.0f ? v : 0.0f;
                    C[(size_t)(rowBase + r) * N + col] = v;
                }
            }
        }
    }
}

// ---------------------------------------------------------------------------
// Per-token "attention over d_k" (faithful to the reference's axis swap):
// for each token, Qm/Km/Vm are [DK=128, H=16]; S = Qm Km^T / sqrt(DK) (+causal
// mask over the dk axes), softmax rows, O = P Vm. One block per token.
// ---------------------------------------------------------------------------
__global__ __launch_bounds__(128) void attn_dk_kernel(
    const float* __restrict__ Q, const float* __restrict__ K,
    const float* __restrict__ V, float* __restrict__ Out, int causal)
{
    __shared__ float Ks[DK * NHEAD];
    __shared__ float Vs[DK * NHEAD];

    const int tok = blockIdx.x;                 // 0..1023
    const float* q = Q + (size_t)tok * D_EMB;
    const float* k = K + (size_t)tok * D_EMB;
    const float* v = V + (size_t)tok * D_EMB;
    const int i = threadIdx.x;                  // dk row 0..127

    float qr[NHEAD];
#pragma unroll
    for (int hh = 0; hh < NHEAD; hh++) {
        qr[hh]             = q[hh * DK + i];
        Ks[i * NHEAD + hh] = k[hh * DK + i];
        Vs[i * NHEAD + hh] = v[hh * DK + i];
    }
    __syncthreads();

    const float scale = 0.08838834764831845f;   // 1/sqrt(128)
    const int jmax = causal ? i : (DK - 1);

    // pass 1: row max
    float m = -3.4e38f;
    for (int j = 0; j <= jmax; j++) {
        float s = 0.0f;
#pragma unroll
        for (int hh = 0; hh < NHEAD; hh++) s += qr[hh] * Ks[j * NHEAD + hh];
        s *= scale;
        m = s > m ? s : m;
    }
    // pass 2: sum of exp + unnormalized output
    float sexp = 0.0f;
    float o[NHEAD];
#pragma unroll
    for (int hh = 0; hh < NHEAD; hh++) o[hh] = 0.0f;
    for (int j = 0; j <= jmax; j++) {
        float s = 0.0f;
#pragma unroll
        for (int hh = 0; hh < NHEAD; hh++) s += qr[hh] * Ks[j * NHEAD + hh];
        const float e = __expf(s * scale - m);
        sexp += e;
#pragma unroll
        for (int hh = 0; hh < NHEAD; hh++) o[hh] += e * Vs[j * NHEAD + hh];
    }
    const float inv = 1.0f / sexp;
    // concat: out[tok, h*DK + i]
#pragma unroll
    for (int hh = 0; hh < NHEAD; hh++)
        Out[(size_t)tok * D_EMB + hh * DK + i] = o[hh] * inv;
}

// ---------------------------------------------------------------------------
// Row LayerNorm over D_EMB=2048. One block (256 threads) per row.
// ---------------------------------------------------------------------------
__global__ __launch_bounds__(256) void layernorm_kernel(
    const float* __restrict__ X, const float* __restrict__ g,
    const float* __restrict__ b, float* __restrict__ Y)
{
    __shared__ float red[256];
    const int row = blockIdx.x;
    const int tid = threadIdx.x;
    const float* x = X + (size_t)row * D_EMB;

    float s = 0.0f;
    for (int d = tid; d < D_EMB; d += 256) s += x[d];
    red[tid] = s; __syncthreads();
    for (int off = 128; off > 0; off >>= 1) {
        if (tid < off) red[tid] += red[tid + off];
        __syncthreads();
    }
    const float mean = red[0] * (1.0f / D_EMB);
    __syncthreads();

    float vs = 0.0f;
    for (int d = tid; d < D_EMB; d += 256) { const float t = x[d] - mean; vs += t * t; }
    red[tid] = vs; __syncthreads();
    for (int off = 128; off > 0; off >>= 1) {
        if (tid < off) red[tid] += red[tid + off];
        __syncthreads();
    }
    const float var = red[0] * (1.0f / D_EMB);
    const float inv = rsqrtf(var + 1e-5f);

    for (int d = tid; d < D_EMB; d += 256)
        Y[(size_t)row * D_EMB + d] = g[d] * (x[d] - mean) * inv + b[d];
}

// ---------------------------------------------------------------------------
// x + sinusoidal positional encoding (computed on the fly).
// ---------------------------------------------------------------------------
__global__ __launch_bounds__(256) void addpe_kernel(
    const float* __restrict__ X, float* __restrict__ Y)
{
    const size_t idx = (size_t)blockIdx.x * 256 + threadIdx.x;   // < 8*128*2048
    const int d = (int)(idx & (D_EMB - 1));
    const int l = (int)((idx >> 11) & (SEQL - 1));
    const int i2 = d & ~1;                                       // 2i
    const float div = __expf(-logf(10000.0f) * (float)i2 / (float)D_EMB);
    const float ang = (float)l * div;
    const float pe = (d & 1) ? __cosf(ang) : __sinf(ang);
    Y[idx] = X[idx] + pe;
}

// ---------------------------------------------------------------------------
// Softmax over the SEQUENCE axis (axis=1 of [B,L,V]) for batch 7 only:
// out[l,v] = exp(lg[l,v]-max_l)/sum_l exp(lg[l,v]-max_l).  lg is [128,30000].
// Threads over v -> fully coalesced per l.
// ---------------------------------------------------------------------------
__global__ __launch_bounds__(256) void colsoftmax_kernel(
    const float* __restrict__ lg, float* __restrict__ out)
{
    const int v = blockIdx.x * 256 + threadIdx.x;
    if (v >= VOCAB) return;
    float m = -3.4e38f;
    for (int l = 0; l < SEQL; l++) {
        const float t = lg[(size_t)l * VOCAB + v];
        m = t > m ? t : m;
    }
    float s = 0.0f;
    for (int l = 0; l < SEQL; l++) s += __expf(lg[(size_t)l * VOCAB + v] - m);
    const float inv = 1.0f / s;
    for (int l = 0; l < SEQL; l++)
        out[(size_t)l * VOCAB + v] = __expf(lg[(size_t)l * VOCAB + v] - m) * inv;
}

// ---------------------------------------------------------------------------
// Host driver
// ---------------------------------------------------------------------------
static inline void launch_gemm(const float* A, const float* B, const float* bias,
                               float* C, int M, int N, int K, int relu,
                               hipStream_t s)
{
    dim3 grid((N + BN - 1) / BN, M / BM);
    gemm_wmma_bf16<<<grid, 256, 0, s>>>(A, B, bias, C, M, N, K, relu);
}

extern "C" void kernel_launch(void* const* d_in, const int* in_sizes, int n_in,
                              void* d_out, int out_size, void* d_ws, size_t ws_size,
                              hipStream_t stream)
{
    (void)in_sizes; (void)n_in; (void)out_size; (void)ws_size;

    const float* x_in  = (const float*)d_in[0];
    const float* ctx   = (const float*)d_in[1];
    const float* Wq1   = (const float*)d_in[2];
    const float* Wk1   = (const float*)d_in[3];
    const float* Wv1   = (const float*)d_in[4];
    const float* Wo1   = (const float*)d_in[5];
    const float* Wq2   = (const float*)d_in[6];
    const float* Wk2   = (const float*)d_in[7];
    const float* Wv2   = (const float*)d_in[8];
    const float* Wo2   = (const float*)d_in[9];
    const float* Wff1  = (const float*)d_in[10];
    const float* bff1  = (const float*)d_in[11];
    const float* Wff2  = (const float*)d_in[12];
    const float* bff2  = (const float*)d_in[13];
    const float* g1    = (const float*)d_in[14];
    const float* be1   = (const float*)d_in[15];
    const float* g2    = (const float*)d_in[16];
    const float* be2   = (const float*)d_in[17];
    const float* g3    = (const float*)d_in[18];
    const float* be3   = (const float*)d_in[19];
    const float* Wlin  = (const float*)d_in[20];
    const float* blin  = (const float*)d_in[21];

    float* ws = (float*)d_ws;
    float* xb = ws;                              // [1024,2048]
    float* yb = xb + (size_t)NTOK * D_EMB;       // [1024,2048]
    float* qb = yb + (size_t)NTOK * D_EMB;       // [1024,2048]
    float* kb = qb + (size_t)NTOK * D_EMB;       // [1024,2048]
    float* vb = kb + (size_t)NTOK * D_EMB;       // [1024,2048]
    float* ab = vb + (size_t)NTOK * D_EMB;       // [1024,2048]
    float* hb = ab + (size_t)NTOK * D_EMB;       // [1024,8192]
    float* lg = hb + (size_t)NTOK * D_FF;        // [128,30000]

    // x = x + PE
    addpe_kernel<<<(NTOK * D_EMB) / 256, 256, 0, stream>>>(x_in, xb);

    for (int layer = 0; layer < 6; layer++) {
        // --- masked self-attention block ---
        launch_gemm(xb, Wq1, nullptr, qb, NTOK, D_EMB, D_EMB, 0, stream);
        launch_gemm(xb, Wk1, nullptr, kb, NTOK, D_EMB, D_EMB, 0, stream);
        launch_gemm(xb, Wv1, nullptr, vb, NTOK, D_EMB, D_EMB, 0, stream);
        attn_dk_kernel<<<NTOK, 128, 0, stream>>>(qb, kb, vb, ab, /*causal=*/1);
        launch_gemm(ab, Wo1, nullptr, yb, NTOK, D_EMB, D_EMB, 0, stream);
        layernorm_kernel<<<NTOK, 256, 0, stream>>>(yb, g1, be1, xb);

        // --- cross-attention block (K/V from raw context) ---
        launch_gemm(xb,  Wq2, nullptr, qb, NTOK, D_EMB, D_EMB, 0, stream);
        launch_gemm(ctx, Wk2, nullptr, kb, NTOK, D_EMB, D_EMB, 0, stream);
        launch_gemm(ctx, Wv2, nullptr, vb, NTOK, D_EMB, D_EMB, 0, stream);
        attn_dk_kernel<<<NTOK, 128, 0, stream>>>(qb, kb, vb, ab, /*causal=*/0);
        launch_gemm(ab, Wo2, nullptr, yb, NTOK, D_EMB, D_EMB, 0, stream);
        layernorm_kernel<<<NTOK, 256, 0, stream>>>(yb, g2, be2, xb);

        // --- feed-forward ---
        launch_gemm(xb, Wff1, bff1, hb, NTOK, D_FF,  D_EMB, 1, stream);
        launch_gemm(hb, Wff2, bff2, yb, NTOK, D_EMB, D_FF,  0, stream);
        layernorm_kernel<<<NTOK, 256, 0, stream>>>(yb, g3, be3, xb);
    }

    // Final projection: only batch 7 survives softmax(axis=1)[-1].
    launch_gemm(xb + (size_t)7 * SEQL * D_EMB, Wlin, blin, lg,
                SEQL, VOCAB, D_EMB, 0, stream);
    colsoftmax_kernel<<<(VOCAB + 255) / 256, 256, 0, stream>>>(lg, (float*)d_out);
}
